// AttractorDynamics_63660005261845
// MI455X (gfx1250) — matrix operand order repaired and checked by reading, not compiled, not run on hardware
//
#include <hip/hip_runtime.h>
#include <math.h>

typedef __attribute__((ext_vector_type(16))) _Float16 v16h;
typedef __attribute__((ext_vector_type(8)))  float    v8f;

#define DT_C       0.05f
#define MAG_FLOOR  1e-3f
#define MAG_CEIL   12.0f
#define D_DIM      1024
#define MB         16      // rows per workgroup
#define WAVES      8
#define TILES      8       // 16-wide N-tiles per wave: 8*16*8 waves = 1024 cols
#define S16_STRIDE 1032    // padded f16 row stride (2064B = 16B-aligned, bank-skewed)

union V16HU { v16h h; float4 f[2]; };

// ---- prep: diffusion f32 -> f16 (row-major, 1024x1024) ----
__global__ void diff_to_f16_kernel(const float* __restrict__ diff,
                                   _Float16* __restrict__ out) {
    int i = (blockIdx.x * blockDim.x + threadIdx.x) * 4;   // 4 elems/thread
    float4 v = *(const float4*)(diff + i);
    union { _Float16 h[4]; float2 p; } u;
    u.h[0] = (_Float16)v.x; u.h[1] = (_Float16)v.y;
    u.h[2] = (_Float16)v.z; u.h[3] = (_Float16)v.w;
    *(float2*)(out + i) = u.p;
}

// ---- main: 20 fused steps, one 16-row block per workgroup ----
__global__ __launch_bounds__(256)
void attractor_kernel(const float* __restrict__ signal,
                      const _Float16* __restrict__ diff16,
                      const float* __restrict__ cubic_scale_p,
                      const int* __restrict__ num_steps_p,
                      float* __restrict__ out) {
    __shared__ _Float16 s16[MB][S16_STRIDE];   // f16 staged A matrix (state)
    __shared__ float rowsum[MB];               // for per-row mean
    __shared__ float rowsq[MB];                // for per-row norm

    const int tid  = threadIdx.x;
    const int lane = tid & 31;
    const int w    = tid >> 5;        // wave id 0..7
    const int n    = lane & 15;       // N within tile / A-row id
    const int g    = lane >> 4;       // lane half
    const int rowBase = blockIdx.x * MB;

    const float cs     = cubic_scale_p[0];
    const int   nsteps = num_steps_p[0];

    // state: st[t][r] holds element (M = r + 8*g, N = w*128 + t*16 + n)
    float st[TILES][8];
    #pragma unroll
    for (int t = 0; t < TILES; ++t)
        #pragma unroll
        for (int r = 0; r < 8; ++r) st[t][r] = 0.0f;

    const v8f vzero = {0.f,0.f,0.f,0.f,0.f,0.f,0.f,0.f};

    for (int step = 0; step < nsteps; ++step) {
        if (tid < MB) { rowsum[tid] = 0.0f; rowsq[tid] = 0.0f; }
        __syncthreads();

        // stage state as f16 A-matrix in LDS + accumulate row sums (for mean)
        float psum[8];
        #pragma unroll
        for (int r = 0; r < 8; ++r) psum[r] = 0.0f;
        #pragma unroll
        for (int t = 0; t < TILES; ++t) {
            const int col = w * 128 + t * 16 + n;
            #pragma unroll
            for (int r = 0; r < 8; ++r) {
                s16[r + 8 * g][col] = (_Float16)st[t][r];
                psum[r] += st[t][r];
            }
        }
        #pragma unroll
        for (int r = 0; r < 8; ++r) atomicAdd(&rowsum[r + 8 * g], psum[r]);
        __syncthreads();

        // ---- GEMM: drift_mm = s @ diffusion^T via v_wmma_f32_16x16x32_f16 ----
        v8f acc[TILES];
        #pragma unroll
        for (int t = 0; t < TILES; ++t) acc[t] = vzero;

        const _Float16* arow = &s16[n][0];     // A row M = n (lanes 0-15 / 16-31)
        for (int k0 = 0; k0 < D_DIM; k0 += 32) {
            // A 16x32 f16 fragment: lane(m=n,g): elems 0..7 = K[k0+g*8 ..],
            //                                    elems 8..15 = K[k0+16+g*8 ..]
            V16HU a;
            a.f[0] = *(const float4*)(arow + k0 + g * 8);
            a.f[1] = *(const float4*)(arow + k0 + 16 + g * 8);
            #pragma unroll
            for (int t = 0; t < TILES; ++t) {
                // B 32x16: B[k][j] = diff[j][k]; lane(g,n) reads
                // diff16 row (j0+n), 8 halves at k0+g*8 and 8 at k0+16+g*8
                const _Float16* brow = diff16
                    + (size_t)(w * 128 + t * 16 + n) * D_DIM + k0 + g * 8;
                V16HU b;
                b.f[0] = *(const float4*)brow;
                b.f[1] = *(const float4*)(brow + 16);
                acc[t] = __builtin_amdgcn_wmma_f32_16x16x32_f16(
                    false, a.h, false, b.h, (short)0, acc[t], false, false);
            }
        }

        // ---- epilogue: cubic + signal + Euler + norm clamp ----
        float vsq[8];
        #pragma unroll
        for (int r = 0; r < 8; ++r) vsq[r] = 0.0f;
        #pragma unroll
        for (int r = 0; r < 8; ++r) {
            const float mean = rowsum[r + 8 * g] * (1.0f / (float)D_DIM);
            const size_t grow = (size_t)(rowBase + r + 8 * g) * D_DIM;
            #pragma unroll
            for (int t = 0; t < TILES; ++t) {
                const int col = w * 128 + t * 16 + n;
                const float sv = st[t][r];
                const float c  = sv - mean;
                const float drift = acc[t][r] + cs * c * c * c + signal[grow + col];
                const float v = sv + DT_C * drift;
                st[t][r] = v;
                vsq[r] += v * v;
            }
        }
        #pragma unroll
        for (int r = 0; r < 8; ++r) atomicAdd(&rowsq[r + 8 * g], vsq[r]);
        __syncthreads();

        #pragma unroll
        for (int r = 0; r < 8; ++r) {
            const float nrm = sqrtf(rowsq[r + 8 * g]);
            float scale;
            if (nrm < MAG_FLOOR)     scale = MAG_FLOOR / (nrm + 1e-15f);
            else if (nrm > MAG_CEIL) scale = MAG_CEIL / nrm;
            else                     scale = 1.0f;
            #pragma unroll
            for (int t = 0; t < TILES; ++t) st[t][r] *= scale;
        }
        __syncthreads();   // protect rowsum/rowsq before next step's zeroing
    }

    // ---- final target-norm rescale + store ----
    if (tid < MB) rowsq[tid] = 0.0f;
    __syncthreads();
    float vsq[8];
    #pragma unroll
    for (int r = 0; r < 8; ++r) vsq[r] = 0.0f;
    #pragma unroll
    for (int r = 0; r < 8; ++r)
        #pragma unroll
        for (int t = 0; t < TILES; ++t) vsq[r] += st[t][r] * st[t][r];
    #pragma unroll
    for (int r = 0; r < 8; ++r) atomicAdd(&rowsq[r + 8 * g], vsq[r]);
    __syncthreads();

    #pragma unroll
    for (int r = 0; r < 8; ++r) {
        const float nrm = sqrtf(rowsq[r + 8 * g]);
        const size_t grow = (size_t)(rowBase + r + 8 * g) * D_DIM;
        if (nrm > 1e-12f) {
            const float inv = 1.0f / nrm;
            #pragma unroll
            for (int t = 0; t < TILES; ++t)
                out[grow + w * 128 + t * 16 + n] = st[t][r] * inv;
        } else {
            #pragma unroll
            for (int t = 0; t < TILES; ++t)
                out[grow + w * 128 + t * 16 + n] = 1.0f / 32.0f;  // 1/sqrt(1024)
        }
    }
}

extern "C" void kernel_launch(void* const* d_in, const int* in_sizes, int n_in,
                              void* d_out, int out_size, void* d_ws, size_t ws_size,
                              hipStream_t stream) {
    const float* signal     = (const float*)d_in[0];   // (16384, 1024) f32
    const float* diffusion  = (const float*)d_in[1];   // (1024, 1024)  f32
    const float* cubic      = (const float*)d_in[2];   // scalar f32
    const int*   num_steps  = (const int*)d_in[3];     // scalar i32
    float*       out        = (float*)d_out;           // (16384, 1024) f32
    _Float16*    diff16     = (_Float16*)d_ws;         // 2 MB scratch

    // diffusion -> f16 (row-major): 1M elems, 4/thread
    diff_to_f16_kernel<<<1024, 256, 0, stream>>>(diffusion, diff16);

    // one workgroup per 16-row block: 16384/16 = 1024 blocks, 8 waves each
    attractor_kernel<<<1024, 256, 0, stream>>>(signal, diff16, cubic, num_steps, out);
}
